// adLIF_81166291959856
// MI455X (gfx1250) — matrix operand order repaired and checked by reading, not compiled, not run on hardware
//
#include <hip/hip_runtime.h>
#include <hip/hip_bf16.h>

// ---------------------------------------------------------------------------
// adLIF spiking RNN on gfx1250 (MI455X).
//
// Strategy:
//  * BatchNorm stats: deterministic 2-pass reduction over (51200, 512).
//  * Wm = W * (1-eye) split into bf16 hi + bf16 residual, pre-packed into the
//    exact WMMA B-fragment lane layout (16x16x32 bf16, wave32).
//  * y (spikes) are exactly {0,1} -> exact in bf16, so
//        y @ Wm  ==  y@W_hi + y@W_lo   (two v_wmma_f32_16x16x32_bf16, f32 acc)
//    recovers near-fp32 accuracy.
//  * 200 sequential step kernels; spikes ping-pong between two buffers, each
//    step writes the next step's A-fragments (bf16, A-layout) directly.
// ---------------------------------------------------------------------------

typedef __attribute__((ext_vector_type(16))) __bf16 v16bf;
typedef __attribute__((ext_vector_type(8)))  float  v8f;

#define Tn 200
#define Bn 256
#define Nn 512
#define TBn (Tn * Bn)          // 51200
#define BNn (Bn * Nn)          // 131072 elements per time slice

#define ALPHA_LO 0.81873075f   // exp(-1/5)
#define ALPHA_HI 0.96078944f   // exp(-1/25)
#define BETA_LO  0.96721610f   // exp(-1/30)
#define BETA_HI  0.99170128f   // exp(-1/120)

__device__ __forceinline__ float clampf(float x, float lo, float hi) {
  return fminf(fmaxf(x, lo), hi);
}

// bf16 round-to-nearest-even of an f32, as raw bits
__device__ __forceinline__ unsigned short f32_to_bf16_bits(float f) {
  unsigned int u = __float_as_uint(f);
  u = u + 0x7FFFu + ((u >> 16) & 1u);
  return (unsigned short)(u >> 16);
}

// ---------------------------------------------------------------------------
// Kernel 1: zero the state region of the workspace (V, WT, Yf[2], blobA[2])
// ---------------------------------------------------------------------------
__global__ void zero_ws(unsigned int* __restrict__ p, int ndw) {
  for (int i = blockIdx.x * blockDim.x + threadIdx.x; i < ndw;
       i += gridDim.x * blockDim.x)
    p[i] = 0u;
}

// ---------------------------------------------------------------------------
// Kernel 2: BN partial sums. 128 blocks: (row-chunk, col-half).
// Each thread owns one column over 800 rows.
// ---------------------------------------------------------------------------
__global__ __launch_bounds__(256) void bn_partial(const float* __restrict__ tx,
                                                  float* __restrict__ ps,
                                                  float* __restrict__ ps2) {
  const int cblk  = blockIdx.x & 1;
  const int chunk = blockIdx.x >> 1;          // 0..63
  const int col   = cblk * 256 + threadIdx.x; // 0..511
  const size_t r0 = (size_t)chunk * 800;
  float s = 0.f, s2 = 0.f;
  for (int r = 0; r < 800; ++r) {
    float x = tx[(r0 + r) * Nn + col];
    s += x;
    s2 += x * x;
  }
  ps[chunk * Nn + col]  = s;
  ps2[chunk * Nn + col] = s2;
}

// ---------------------------------------------------------------------------
// Kernel 3: finalize BN stats + build per-neuron parameter table.
// prm layout (floats): [0:512) scale, [512) shift, [1024) alpha, [1536) beta,
//                      [2048) a, [2560) b
// ---------------------------------------------------------------------------
__global__ __launch_bounds__(256) void bn_finalize(
    const float* __restrict__ ps, const float* __restrict__ ps2,
    const float* __restrict__ alpha, const float* __restrict__ beta,
    const float* __restrict__ a, const float* __restrict__ b,
    const float* __restrict__ gamma, const float* __restrict__ bnb,
    float* __restrict__ prm) {
  const int col = blockIdx.x * 256 + threadIdx.x; // grid = 2
  float s = 0.f, s2 = 0.f;
  for (int c = 0; c < 64; ++c) {   // fixed order -> deterministic
    s += ps[c * Nn + col];
    s2 += ps2[c * Nn + col];
  }
  const float inv = 1.0f / (float)TBn;
  const float mean = s * inv;
  const float var  = s2 * inv - mean * mean;   // biased variance
  const float invstd = rsqrtf(var + 1e-5f);
  const float scale  = gamma[col] * invstd;
  prm[col]            = scale;
  prm[Nn + col]       = bnb[col] - mean * scale;
  prm[2 * Nn + col]   = clampf(alpha[col], ALPHA_LO, ALPHA_HI);
  prm[3 * Nn + col]   = clampf(beta[col], BETA_LO, BETA_HI);
  prm[4 * Nn + col]   = clampf(a[col], 0.f, 1.f);
  prm[5 * Nn + col]   = clampf(b[col], 0.f, 1.f);
}

// ---------------------------------------------------------------------------
// Kernel 4: pack Wm (diag zeroed) into bf16 hi/lo B-fragment layout.
// B fragment (16x16x32 bf16, wave32): for lane L, vgpr r (0..7):
//   column N = L&15,  K = (L>=16 ? 8 : 0) + (r>=4 ? 16 : 0) + 2(r&3) + halfword
// Blob layout (shorts): [ntile][chunk][lane][16]  (ntile=N/16, chunk=K/32)
// ---------------------------------------------------------------------------
__global__ __launch_bounds__(256) void pack_W(const float* __restrict__ W,
                                              unsigned short* __restrict__ bhi,
                                              unsigned short* __restrict__ blo) {
  const int i = blockIdx.x * 256 + threadIdx.x; // 0..262143, [k][n] row-major
  const int k = i >> 9, n = i & (Nn - 1);
  float w = (k == n) ? 0.f : W[i];
  unsigned short hb = f32_to_bf16_bits(w);
  float whi = __uint_as_float(((unsigned int)hb) << 16);
  unsigned short lb = f32_to_bf16_bits(w - whi);

  const int c  = k >> 5, kk = k & 31;
  const int nt = n >> 4, nl = n & 15;
  const int L  = nl + ((kk & 8) ? 16 : 0);
  const int r  = ((kk >= 16) ? 4 : 0) + ((kk & 7) >> 1);
  const int h  = kk & 1;
  const size_t di = (((size_t)(nt * 16 + c) * 32 + L) * 8 + r) * 2 + h;
  bhi[di] = hb;
  blo[di] = lb;
}

// ---------------------------------------------------------------------------
// Kernel 5: one adLIF time step.
// Grid: (4 n-groups, 16 m-blocks), 256 threads = 8 waves.
// Each wave computes one 16x16 tile of  rec = y_prev @ Wm  via bf16 WMMA
// (hi+lo split, f32 accumulate, K=512 in 16 chunks), then the elementwise
// adLIF update + spike, and scatters the new spikes into next step's
// A-fragment blob (bf16) as well as fp32 state/out.
//
// A fragment (16x32 bf16): lane L<16 holds row M=L, K = {2r,2r+1} (r<4),
// {16+2(r-4), ..} (r>=4); lane L>=16 same rows with K offset +8 / +24.
// A blob (shorts): [mblk][chunk][lane][16]
// ---------------------------------------------------------------------------
__global__ __launch_bounds__(256) void adlif_step(
    const float* __restrict__ tx_t,            // tx + t*B*N
    const float* __restrict__ prm,
    const unsigned short* __restrict__ bhi,
    const unsigned short* __restrict__ blo,
    const unsigned short* __restrict__ blobA_in,
    unsigned short* __restrict__ blobA_out,
    const float* __restrict__ yf_in,
    float* __restrict__ yf_out,
    float* __restrict__ V,
    float* __restrict__ WT,
    float* __restrict__ out_t)                 // d_out + t*B*N
{
  __shared__ unsigned int ldsA[4096];          // 16KB: [chunk][lane][8 dwords]
  const int tid  = threadIdx.x;
  const int lane = tid & 31;
  const int wv   = tid >> 5;
  const int mblk = blockIdx.y;                 // 0..15 (batch / 16)
  const int ntile = blockIdx.x * 8 + wv;       // 0..31 (neuron / 16)

  // Stage this m-block's A fragments (y_prev, bf16 packed) into LDS.
  {
    const uint4* __restrict__ src =
        (const uint4*)(blobA_in + (size_t)mblk * 8192); // 8192 shorts = 1024 uint4
    uint4* dst = (uint4*)ldsA;
#pragma unroll
    for (int i = 0; i < 4; ++i) dst[tid + i * 256] = src[tid + i * 256];
  }
  __syncthreads();

  v8f acc = {0.f, 0.f, 0.f, 0.f, 0.f, 0.f, 0.f, 0.f};
  const unsigned short* __restrict__ bhp = bhi + (size_t)ntile * 16 * 32 * 16;
  const unsigned short* __restrict__ blp = blo + (size_t)ntile * 16 * 32 * 16;

#pragma unroll 4
  for (int c = 0; c < 16; ++c) {
    const v16bf av = *(const v16bf*)&ldsA[(c * 32 + lane) * 8];
    const v16bf bh = *(const v16bf*)(bhp + (size_t)(c * 32 + lane) * 16);
    const v16bf bl = *(const v16bf*)(blp + (size_t)(c * 32 + lane) * 16);
    if (c + 1 < 16)
      __builtin_prefetch(bhp + (size_t)((c + 1) * 32 + lane) * 16, 0, 0);
    acc = __builtin_amdgcn_wmma_f32_16x16x32_bf16(false, av, false, bh,
                                                  (short)0, acc, false, false);
    acc = __builtin_amdgcn_wmma_f32_16x16x32_bf16(false, av, false, bl,
                                                  (short)0, acc, false, false);
  }

  // Elementwise adLIF update. C layout: vgpr r -> M = r + 8*(lane>=16),
  // N = lane&15.
  const int nl = lane & 15;
  const int Ng = ntile * 16 + nl;
  const float scale = prm[Ng];
  const float shift = prm[Nn + Ng];
  const float al    = prm[2 * Nn + Ng];
  const float be    = prm[3 * Nn + Ng];
  const float aa    = prm[4 * Nn + Ng];
  const float bb    = prm[5 * Nn + Ng];

  // Scatter constants for writing y_new into next step's A blob.
  const int cN    = Ng >> 5;
  const int kk    = Ng & 31;
  const int r4    = (kk >= 16) ? 4 : 0;
  const int base  = kk & 15;
  const int lplus = (base >= 8) ? 16 : 0;
  const int rp    = r4 + ((base & 7) >> 1);
  const int hbit  = kk & 1;
  const int mhi   = (lane >> 4) * 8;

#pragma unroll
  for (int r = 0; r < 8; ++r) {
    const int Mloc = r + mhi;
    const int Mg   = mblk * 16 + Mloc;
    const size_t idx = (size_t)Mg * Nn + Ng;

    const float x  = tx_t[idx] * scale + shift + acc[r]; // BN + recurrent
    const float vo = V[idx];
    const float wo = WT[idx];
    const float yo = yf_in[idx];

    const float wn = be * wo + aa * vo + bb * yo;
    const float vn = al * (vo - yo) + (1.0f - al) * (x - wn);
    const float yn = (vn > 0.3f) ? 1.0f : 0.0f;

    V[idx]      = vn;
    WT[idx]     = wn;
    yf_out[idx] = yn;
    out_t[idx]  = yn;

    const size_t di =
        (((size_t)(mblk * 16 + cN) * 32 + (Mloc + lplus)) * 8 + rp) * 2 + hbit;
    blobA_out[di] = (vn > 0.3f) ? (unsigned short)0x3F80 : (unsigned short)0;
  }
}

// ---------------------------------------------------------------------------
// Workspace layout (bytes):
//   V       @ 0        524288
//   WT      @ 524288   524288
//   Yf0     @ 1048576  524288
//   Yf1     @ 1572864  524288
//   blobA0  @ 2097152  262144
//   blobA1  @ 2359296  262144
//   Bhi     @ 2621440  524288
//   Blo     @ 3145728  524288
//   ps      @ 3670016  131072
//   ps2     @ 3801088  131072
//   prm     @ 3932160  12288
// ---------------------------------------------------------------------------
extern "C" void kernel_launch(void* const* d_in, const int* in_sizes, int n_in,
                              void* d_out, int out_size, void* d_ws,
                              size_t ws_size, hipStream_t stream) {
  const float* tx    = (const float*)d_in[0];
  const float* alpha = (const float*)d_in[1];
  const float* beta  = (const float*)d_in[2];
  const float* a     = (const float*)d_in[3];
  const float* b     = (const float*)d_in[4];
  const float* gamma = (const float*)d_in[5];
  const float* bnb   = (const float*)d_in[6];
  const float* W     = (const float*)d_in[7];
  float* out = (float*)d_out;

  char* ws = (char*)d_ws;
  float* V  = (float*)(ws + 0);
  float* WT = (float*)(ws + 524288);
  float* Yf[2] = {(float*)(ws + 1048576), (float*)(ws + 1572864)};
  unsigned short* blobA[2] = {(unsigned short*)(ws + 2097152),
                              (unsigned short*)(ws + 2359296)};
  unsigned short* Bhi = (unsigned short*)(ws + 2621440);
  unsigned short* Blo = (unsigned short*)(ws + 3145728);
  float* ps  = (float*)(ws + 3670016);
  float* ps2 = (float*)(ws + 3801088);
  float* prm = (float*)(ws + 3932160);

  // Zero V, WT, Yf[2], blobA[2]  (bytes [0, 2621440) = 655360 dwords)
  zero_ws<<<640, 256, 0, stream>>>((unsigned int*)ws, 655360);

  bn_partial<<<128, 256, 0, stream>>>(tx, ps, ps2);
  bn_finalize<<<2, 256, 0, stream>>>(ps, ps2, alpha, beta, a, b, gamma, bnb,
                                     prm);
  pack_W<<<1024, 256, 0, stream>>>(W, Bhi, Blo);

  for (int t = 0; t < Tn; ++t) {
    const int pi = t & 1, po = pi ^ 1;
    adlif_step<<<dim3(4, 16), 256, 0, stream>>>(
        tx + (size_t)t * BNn, prm, Bhi, Blo, blobA[pi], blobA[po], Yf[pi],
        Yf[po], V, WT, out + (size_t)t * BNn);
  }
}
